// GroupedQueryAttention_70703751627570
// MI455X (gfx1250) — matrix operand order repaired and checked by reading, not compiled, not run on hardware
//
#include <hip/hip_runtime.h>

#define B_  8
#define S_  512
#define D_  4096
#define G_  32
#define DH_ 128
#define RD_ 1024

typedef __attribute__((ext_vector_type(16))) __bf16 v16bf;
typedef __attribute__((ext_vector_type(8)))  __bf16 v8bf;
typedef __attribute__((ext_vector_type(8)))  float  v8f;
typedef __attribute__((ext_vector_type(16))) float  v16f;

__device__ inline __bf16 f2bf(float f) {
  union { float f; unsigned u; } v; v.f = f;
  unsigned r = v.u + 0x7FFFu + ((v.u >> 16) & 1u);   // round-to-nearest-even
  unsigned short h = (unsigned short)(r >> 16);
  __bf16 o; __builtin_memcpy(&o, &h, 2); return o;
}

// -------------------------------------------------------------------------
// Elementwise f32 -> bf16 cast, 8 elements/thread (bandwidth-bound).
// -------------------------------------------------------------------------
__global__ __launch_bounds__(256) void cast_bf16_kernel(
    const float* __restrict__ in, __bf16* __restrict__ out)
{
  const size_t i8 = ((size_t)blockIdx.x * 256 + threadIdx.x) * 8;
  const v8f f = *(const v8f*)(in + i8);
  v8bf o;
#pragma unroll
  for (int i = 0; i < 8; ++i) o[i] = f2bf(f[i]);
  *(v8bf*)(out + i8) = o;
}

// -------------------------------------------------------------------------
// GEMM: Out[M,N] = A[M,K] @ W[N,K]^T  (y = x @ W.T, W row-major [N,K]).
// bf16 WMMA, f32 accumulate. A/W either bf16 (fast path: zero conversion
// VALU in the hot loop) or f32 converted on the fly (workspace fallback).
// Block = 256 threads = 8 waves (4x2), block tile 128x128, wave tile 32x64.
// -------------------------------------------------------------------------
template<bool A_BF16, bool W_BF16, bool OUT_BF16>
__global__ __launch_bounds__(256) void gemm_xwT(
    const void* __restrict__ Ap, const void* __restrict__ Wp,
    void* __restrict__ Out, int M, int N, int K)
{
  const int lane = threadIdx.x & 31;
  const int wave = threadIdx.x >> 5;
  const int wm = wave >> 1, wn = wave & 1;
  const int half = lane >> 4, l16 = lane & 15;
  const int rowBase = blockIdx.y * 128 + wm * 32;
  const int colBase = blockIdx.x * 128 + wn * 64;

  v8f acc[2][4];
#pragma unroll
  for (int i = 0; i < 2; ++i)
#pragma unroll
    for (int j = 0; j < 4; ++j) acc[i][j] = (v8f){0,0,0,0,0,0,0,0};

  const __bf16* Ab = (const __bf16*)Ap;
  const float*  Af = (const float*)Ap;
  const __bf16* Wb = (const __bf16*)Wp;
  const float*  Wf = (const float*)Wp;

  for (int kk = 0; kk < K; kk += 32) {
    // A fragments (16x32: lanes 0-15 hold M=0-15 / K{0-15}; lanes 16-31 K{16-31})
    v16bf a[2];
#pragma unroll
    for (int mt = 0; mt < 2; ++mt) {
      const size_t row = (size_t)(rowBase + mt * 16 + l16);
      const size_t kb  = (size_t)(kk + half * 16);
      if (A_BF16) {
        v8bf c0 = *(const v8bf*)(Ab + row * K + kb);
        v8bf c1 = *(const v8bf*)(Ab + row * K + kb + 8);
        a[mt] = __builtin_shufflevector(c0, c1, 0,1,2,3,4,5,6,7,8,9,10,11,12,13,14,15);
      } else {
        v8f f0 = *(const v8f*)(Af + row * K + kb);
        v8f f1 = *(const v8f*)(Af + row * K + kb + 8);
#pragma unroll
        for (int i = 0; i < 8; ++i) { a[mt][i] = f2bf(f0[i]); a[mt][i + 8] = f2bf(f1[i]); }
      }
    }
    // B fragments (32x16): lane = output column n = row n of W; 16 contiguous K
    v16bf bfr[4];
#pragma unroll
    for (int nt = 0; nt < 4; ++nt) {
      const size_t n = (size_t)(colBase + nt * 16 + l16);
      if (W_BF16) {
        bfr[nt] = *(const v16bf*)(Wb + n * (size_t)K + kk + half * 16);
      } else {
        const v16f wv = *(const v16f*)(Wf + n * (size_t)K + kk + half * 16);
#pragma unroll
        for (int i = 0; i < 16; ++i) bfr[nt][i] = f2bf(wv[i]);
      }
    }
    // Prefetch next k-tile (lowers to global_prefetch_b8 on gfx1250)
    if (kk + 32 < K) {
      const size_t row = (size_t)(rowBase + l16);
      const size_t n   = (size_t)(colBase + l16);
      if (A_BF16) __builtin_prefetch(Ab + row * K + kk + 32, 0, 0);
      else        __builtin_prefetch(Af + row * K + kk + 32, 0, 0);
      if (W_BF16) __builtin_prefetch(Wb + n * (size_t)K + kk + 32, 0, 0);
      else        __builtin_prefetch(Wf + n * (size_t)K + kk + 32, 0, 0);
    }
#pragma unroll
    for (int mt = 0; mt < 2; ++mt)
#pragma unroll
      for (int nt = 0; nt < 4; ++nt)
        acc[mt][nt] = __builtin_amdgcn_wmma_f32_16x16x32_bf16(
            false, a[mt], false, bfr[nt], (short)0, acc[mt][nt], false, false);
  }

  // C layout: vgpr r, lanes 0-15 -> M=r, lanes 16-31 -> M=r+8; N = lane%16
#pragma unroll
  for (int mt = 0; mt < 2; ++mt)
#pragma unroll
    for (int nt = 0; nt < 4; ++nt)
#pragma unroll
      for (int r = 0; r < 8; ++r) {
        const size_t row = (size_t)(rowBase + mt * 16 + r + half * 8);
        const size_t col = (size_t)(colBase + nt * 16 + l16);
        if (OUT_BF16) ((__bf16*)Out)[row * (size_t)N + col] = f2bf(acc[mt][nt][r]);
        else          ((float*)Out)[row * (size_t)N + col]  = acc[mt][nt][r];
      }
}

// -------------------------------------------------------------------------
// RoPE: one scalar angle per token (S == RD/2 makes pos*inv_freq elementwise
// in the reference) applied to first RD dims of q and k (bf16 in-place).
// -------------------------------------------------------------------------
__global__ __launch_bounds__(256) void rope_kernel(
    __bf16* __restrict__ q, __bf16* __restrict__ k, const float* __restrict__ pos)
{
  const int token = blockIdx.x;              // b*S + s
  const int s = token & (S_ - 1);
  const float ang = pos[token] *
      __expf(-(2.0f * (float)s / (float)RD_) * 9.210340371976184f); // ln(10000)
  const float c = __cosf(ang), sn = __sinf(ang);
  for (int j = threadIdx.x; j < RD_ / 2; j += 256) {
    const size_t i0 = (size_t)token * D_ + j;
    const size_t i1 = i0 + RD_ / 2;
    {
      const float x1 = (float)q[i0], x2 = (float)q[i1];
      q[i0] = f2bf(x1 * c - x2 * sn);
      q[i1] = f2bf(x2 * c + x1 * sn);
    }
    {
      const float x1 = (float)k[i0], x2 = (float)k[i1];
      k[i0] = f2bf(x1 * c - x2 * sn);
      k[i1] = f2bf(x2 * c + x1 * sn);
    }
  }
}

// -------------------------------------------------------------------------
// V transpose: v[b,s,g*DH+dh] -> vT[(b*G+g)*DH+dh, s]  (per-head DHxS)
// -------------------------------------------------------------------------
__global__ __launch_bounds__(256) void vtrans_kernel(
    const __bf16* __restrict__ v, __bf16* __restrict__ vT)
{
  const size_t idx = (size_t)blockIdx.x * 256 + threadIdx.x;   // < B*S*D
  const int d = (int)(idx & (D_ - 1));
  const int t = (int)(idx >> 12);      // / D
  const int s = t & (S_ - 1);
  const int b = t >> 9;                // / S
  const int g = d >> 7;                // / DH
  const int dh = d & (DH_ - 1);
  vT[(((size_t)(b * G_ + g)) * DH_ + dh) * S_ + s] = v[idx];
}

// -------------------------------------------------------------------------
// Flash attention, WMMA both GEMMs. Wave = 16 query rows; 8 waves/block
// cover 128 queries; grid = B*G*(S/128). Online softmax in C-layout;
// P staged through per-wave LDS to re-shape into A-fragment layout.
// -------------------------------------------------------------------------
__global__ __launch_bounds__(256) void attn_kernel(
    const __bf16* __restrict__ q, const __bf16* __restrict__ k,
    const __bf16* __restrict__ vT, const int* __restrict__ mask,
    __bf16* __restrict__ attn)
{
  const int qblk = blockIdx.x & 3;
  const int bg   = blockIdx.x >> 2;
  const int b    = bg >> 5;
  const int g    = bg & 31;
  const int lane = threadIdx.x & 31;
  const int wave = threadIdx.x >> 5;
  const int half = lane >> 4, l16 = lane & 15;
  const int qbase = qblk * 128 + wave * 16;
  const float SCALE = 0.08838834764831845f;   // 1/sqrt(128)

  __shared__ __bf16 p_lds[8][16 * 64];

  // Q fragments: 4 K-tiles of 32 along DH=128
  v16bf aq[4];
  {
    const __bf16* qrow = q + ((size_t)(b * S_) + qbase + l16) * D_ + g * DH_;
#pragma unroll
    for (int t = 0; t < 4; ++t) {
      const int kb = t * 32 + half * 16;
      v8bf c0 = *(const v8bf*)(qrow + kb);
      v8bf c1 = *(const v8bf*)(qrow + kb + 8);
      aq[t] = __builtin_shufflevector(c0, c1, 0,1,2,3,4,5,6,7,8,9,10,11,12,13,14,15);
    }
  }

  float m_r[8], l_r[8];
  v8f o[8];
#pragma unroll
  for (int r = 0; r < 8; ++r) { m_r[r] = -3.0e38f; l_r[r] = 0.0f; }
#pragma unroll
  for (int ot = 0; ot < 8; ++ot) o[ot] = (v8f){0,0,0,0,0,0,0,0};

  for (int kb0 = 0; kb0 < S_; kb0 += 64) {
    // ---- scores: 16 queries x 64 keys via 4x4 WMMA ----
    v8f sc[4];
#pragma unroll
    for (int nt = 0; nt < 4; ++nt) {
      const int key = kb0 + nt * 16 + l16;
      const __bf16* krow = k + ((size_t)(b * S_) + key) * D_ + g * DH_;
      v8f a = (v8f){0,0,0,0,0,0,0,0};
#pragma unroll
      for (int t = 0; t < 4; ++t) {
        v16bf bk = *(const v16bf*)(krow + t * 32 + half * 16);
        a = __builtin_amdgcn_wmma_f32_16x16x32_bf16(
            false, aq[t], false, bk, (short)0, a, false, false);
      }
      const float bias = (mask[b * S_ + key] == 0) ? -1.0e30f : 0.0f;
#pragma unroll
      for (int r = 0; r < 8; ++r) sc[nt][r] = a[r] * SCALE + bias;
    }

    // ---- online softmax: row reductions over 4 tiles + 16-lane butterfly ----
    float alpha[8];
#pragma unroll
    for (int r = 0; r < 8; ++r) {
      float mx = fmaxf(fmaxf(sc[0][r], sc[1][r]), fmaxf(sc[2][r], sc[3][r]));
      mx = fmaxf(mx, __shfl_xor(mx, 1, 32));
      mx = fmaxf(mx, __shfl_xor(mx, 2, 32));
      mx = fmaxf(mx, __shfl_xor(mx, 4, 32));
      mx = fmaxf(mx, __shfl_xor(mx, 8, 32));
      const float mn = fmaxf(m_r[r], mx);
      alpha[r] = __expf(m_r[r] - mn);
      m_r[r] = mn;
    }
#pragma unroll
    for (int r = 0; r < 8; ++r) {
      float rs = 0.0f;
#pragma unroll
      for (int nt = 0; nt < 4; ++nt) {
        const float p = __expf(sc[nt][r] - m_r[r]);
        sc[nt][r] = p;
        rs += p;
      }
      rs += __shfl_xor(rs, 1, 32);
      rs += __shfl_xor(rs, 2, 32);
      rs += __shfl_xor(rs, 4, 32);
      rs += __shfl_xor(rs, 8, 32);
      l_r[r] = l_r[r] * alpha[r] + rs;
    }
#pragma unroll
    for (int ot = 0; ot < 8; ++ot)
#pragma unroll
      for (int r = 0; r < 8; ++r) o[ot][r] *= alpha[r];

    // ---- P (C-layout) -> LDS row-major -> A-fragment layout ----
    __bf16* pw = &p_lds[wave][0];
#pragma unroll
    for (int nt = 0; nt < 4; ++nt)
#pragma unroll
      for (int r = 0; r < 8; ++r)
        pw[(r + half * 8) * 64 + nt * 16 + l16] = f2bf(sc[nt][r]);

    v16bf ap[2];
#pragma unroll
    for (int ks = 0; ks < 2; ++ks) {
      const __bf16* pr = pw + l16 * 64 + ks * 32 + half * 16;
      v8bf c0 = *(const v8bf*)pr;
      v8bf c1 = *(const v8bf*)(pr + 8);
      ap[ks] = __builtin_shufflevector(c0, c1, 0,1,2,3,4,5,6,7,8,9,10,11,12,13,14,15);
    }

    // ---- O += P @ V : 8 DH-tiles x 2 K-steps ----
#pragma unroll
    for (int ot = 0; ot < 8; ++ot) {
      const __bf16* vrow = vT + ((size_t)bg * DH_ + ot * 16 + l16) * S_ + kb0;
#pragma unroll
      for (int ks = 0; ks < 2; ++ks) {
        v16bf bv = *(const v16bf*)(vrow + ks * 32 + half * 16);
        o[ot] = __builtin_amdgcn_wmma_f32_16x16x32_bf16(
            false, ap[ks], false, bv, (short)0, o[ot], false, false);
      }
    }
  }

  // ---- epilogue: normalize and store bf16 attn[b, q, g*DH + dh] ----
#pragma unroll
  for (int ot = 0; ot < 8; ++ot)
#pragma unroll
    for (int r = 0; r < 8; ++r) {
      const size_t row = (size_t)(b * S_) + qbase + r + half * 8;
      const size_t col = (size_t)g * DH_ + ot * 16 + l16;
      const float val = l_r[r] > 0.0f ? o[ot][r] / l_r[r] : 0.0f;
      attn[row * D_ + col] = f2bf(val);
    }
}

// -------------------------------------------------------------------------
extern "C" void kernel_launch(void* const* d_in, const int* in_sizes, int n_in,
                              void* d_out, int out_size, void* d_ws, size_t ws_size,
                              hipStream_t stream) {
  const float* query = (const float*)d_in[0];
  const float* key   = (const float*)d_in[1];
  const float* value = (const float*)d_in[2];
  const int*   mask  = (const int*)d_in[3];
  const float* pos   = (const float*)d_in[4];
  const float* Wq    = (const float*)d_in[5];
  const float* Wk    = (const float*)d_in[6];
  const float* Wv    = (const float*)d_in[7];
  const float* Wo    = (const float*)d_in[8];
  float* out = (float*)d_out;

  const size_t NE = (size_t)B_ * S_ * D_;   // 16.78M elems (== D*D)
  __bf16* qb = (__bf16*)d_ws;               // bf16 q/k/v after projection
  __bf16* kb = qb + NE;
  __bf16* vb = kb + NE;
  __bf16* vT = vb + NE;
  __bf16* ab = vT + NE;                     // attention output, bf16

  dim3 blk(256);
  dim3 gg(D_ / 128, (B_ * S_) / 128);
  const unsigned castGrid = (unsigned)(NE / (256 * 8));

  if (ws_size >= 12 * NE * sizeof(__bf16)) {
    // Fast path: pre-cast activations + weights to bf16 so the GEMM hot loop
    // is pure b128 loads + WMMA (no conversion VALU competing with XDL).
    __bf16* xq = ab + NE;
    __bf16* xk = xq + NE;
    __bf16* xv = xk + NE;
    __bf16* wq = xv + NE;
    __bf16* wk = wq + NE;
    __bf16* wv = wk + NE;
    __bf16* wo = wv + NE;

    cast_bf16_kernel<<<castGrid, blk, 0, stream>>>(query, xq);
    cast_bf16_kernel<<<castGrid, blk, 0, stream>>>(key,   xk);
    cast_bf16_kernel<<<castGrid, blk, 0, stream>>>(value, xv);
    cast_bf16_kernel<<<castGrid, blk, 0, stream>>>(Wq, wq);
    cast_bf16_kernel<<<castGrid, blk, 0, stream>>>(Wk, wk);
    cast_bf16_kernel<<<castGrid, blk, 0, stream>>>(Wv, wv);
    cast_bf16_kernel<<<castGrid, blk, 0, stream>>>(Wo, wo);

    gemm_xwT<true, true, true><<<gg, blk, 0, stream>>>(xq, wq, qb, B_ * S_, D_, D_);
    gemm_xwT<true, true, true><<<gg, blk, 0, stream>>>(xk, wk, kb, B_ * S_, D_, D_);
    gemm_xwT<true, true, true><<<gg, blk, 0, stream>>>(xv, wv, vb, B_ * S_, D_, D_);
    rope_kernel<<<dim3(B_ * S_), blk, 0, stream>>>(qb, kb, pos);
    vtrans_kernel<<<dim3((unsigned)(NE / 256)), blk, 0, stream>>>(vb, vT);
    attn_kernel<<<dim3(B_ * G_ * (S_ / 128)), blk, 0, stream>>>(qb, kb, vT, mask, ab);
    gemm_xwT<true, true, false><<<gg, blk, 0, stream>>>(ab, wo, out, B_ * S_, D_, D_);
  } else {
    // Fallback: convert f32 -> bf16 on the fly inside the GEMM (L2-resident
    // weights make the extra f32 traffic cheap; costs conversion VALU).
    gemm_xwT<false, false, true><<<gg, blk, 0, stream>>>(query, Wq, qb, B_ * S_, D_, D_);
    gemm_xwT<false, false, true><<<gg, blk, 0, stream>>>(key,   Wk, kb, B_ * S_, D_, D_);
    gemm_xwT<false, false, true><<<gg, blk, 0, stream>>>(value, Wv, vb, B_ * S_, D_, D_);
    rope_kernel<<<dim3(B_ * S_), blk, 0, stream>>>(qb, kb, pos);
    vtrans_kernel<<<dim3((unsigned)(NE / 256)), blk, 0, stream>>>(vb, vT);
    attn_kernel<<<dim3(B_ * G_ * (S_ / 128)), blk, 0, stream>>>(qb, kb, vT, mask, ab);
    gemm_xwT<true, false, false><<<gg, blk, 0, stream>>>(ab, Wo, out, B_ * S_, D_, D_);
  }
}
